// SO3Linear_13125420056868
// MI455X (gfx1250) — compile-verified, matrix operand order checked by reading, not run on hardware
//
#include <hip/hip_runtime.h>

// ---------------------------------------------------------------------------
// SO3 equivariant linear layer, fused for MI455X (gfx1250, wave32, WMMA).
//
//   inter[b,g,i] = sum_{z in group g} CG[z]*sh[b,M2[z]] * x[b,M1[z],i]
//   out[b,seg2[g],o] += inter[b,g,:] @ weight[l_ind[g]]
//
// GEMM (21 GFLOP) on the matrix pipe: v_wmma_f32_16x16x32_bf16 with a hi/lo
// bf16 split (3 products) for ~f32 accuracy. 64-row batch tile = 4 WMMA
// M-tiles per block so every weight fragment read from L2 feeds 4 M-tiles
// (4x lower B traffic, 24 independent WMMAs between load batches).
// ---------------------------------------------------------------------------

typedef __attribute__((ext_vector_type(16))) __bf16 v16bf;
typedef __attribute__((ext_vector_type(8)))  __bf16 v8bf;
typedef __attribute__((ext_vector_type(8)))  float  v8f;
typedef __attribute__((ext_vector_type(4)))  float  v4f;

#define CI 256
#define CO 256
#define NORD 16
#define BT 64          // batch rows per block
#define MT 4           // WMMA M-tiles per block (BT/16)
#define A_STRIDE 264   // padded bf16 row stride in LDS (528 B -> conflict-light)

// ---- prep kernel 1: contiguous range starts from sorted seg1/seg2 ----------
__global__ void so3_build_ranges(const int* __restrict__ seg1, int nnz,
                                 const int* __restrict__ seg2, int G,
                                 int* __restrict__ gStart, int* __restrict__ zStart) {
  int t = threadIdx.x;
  if (t <= NORD) {                 // gStart[m] = #groups with seg2 < m
    int c = 0;
    for (int g = 0; g < G; ++g) c += (seg2[g] < t) ? 1 : 0;
    gStart[t] = c;
  }
  for (int g = t; g <= G; g += blockDim.x) {  // zStart[g] = #nnz with seg1 < g
    int c = 0;
    for (int z = 0; z < nnz; ++z) c += (seg1[z] < g) ? 1 : 0;
    zStart[g] = c;
  }
}

// ---- prep kernel 2: transpose weight to o-major, split to bf16 hi/lo -------
__global__ void so3_split_weight(const float* __restrict__ W, int total,
                                 __bf16* __restrict__ WhiT, __bf16* __restrict__ WloT) {
  int e = blockIdx.x * blockDim.x + threadIdx.x;
  if (e >= total) return;
  int w   = e >> 16;
  int rem = e & 65535;
  int o   = rem >> 8;
  int i   = rem & 255;
  float v  = W[((size_t)w << 16) + (i << 8) + o];  // weight[w][i][o]
  __bf16 h = (__bf16)v;
  WhiT[e]  = h;                                    // WhiT[w][o][i]
  WloT[e]  = (__bf16)(v - (float)h);
}

// ---- main fused kernel -----------------------------------------------------
__global__ __launch_bounds__(256)
void so3_fused(const float* __restrict__ x, const float* __restrict__ sh,
               const float* __restrict__ CG,
               const int* __restrict__ M1, const int* __restrict__ M2,
               const int* __restrict__ l_ind,
               const int* __restrict__ gStart, const int* __restrict__ zStart,
               const __bf16* __restrict__ WhiT, const __bf16* __restrict__ WloT,
               float* __restrict__ out)
{
  __shared__ __bf16 Ahi[BT * A_STRIDE];
  __shared__ __bf16 Alo[BT * A_STRIDE];

  const int tid  = threadIdx.x;
  const int lane = tid & 31;
  const int wave = tid >> 5;          // 8 waves; wave w owns N-columns [32w, 32w+32)
  const int Mord = blockIdx.y;        // output order (0..15)
  const int b0   = blockIdx.x * BT;   // batch tile base

  // inter-builder indexing: per pass, thread -> (batch row, 16-float ci segment)
  const int rr = tid >> 4;            // row within 16-row pass
  const int c0 = (tid & 15) << 4;

  // WMMA fragment indexing (ISA 7.12.2): lane = half*16 + l15
  const int half = lane >> 4;
  const int l15  = lane & 15;

  v8f acc[MT][2] = {};

  const int gs = gStart[Mord];
  const int ge = gStart[Mord + 1];

  for (int g = gs; g < ge; ++g) {
    const int zs = zStart[g], zl = zStart[g + 1];

    // ---- build inter tile (64 x 256) for this group: 4 passes of 16 rows ---
    #pragma unroll
    for (int p = 0; p < MT; ++p) {
      const int r = p * 16 + rr;
      const int b = b0 + r;
      v4f a0 = {}, a1 = {}, a2 = {}, a3 = {};
      for (int z = zs; z < zl; ++z) {
        const int m1 = M1[z], m2 = M2[z];
        const float coef = CG[z] * sh[b * NORD + m2];
        const v4f* xv = (const v4f*)(x + (((size_t)b * NORD + m1) << 8) + c0);
        a0 += coef * xv[0];
        a1 += coef * xv[1];
        a2 += coef * xv[2];
        a3 += coef * xv[3];
      }
      v4f vv0[2] = {a0, a1};
      v4f vv1[2] = {a2, a3};
      v8bf h0, l0, h1, l1;
      #pragma unroll
      for (int q = 0; q < 2; ++q)
        #pragma unroll
        for (int j = 0; j < 4; ++j) {
          float v  = vv0[q][j];
          __bf16 h = (__bf16)v;
          h0[q * 4 + j] = h;
          l0[q * 4 + j] = (__bf16)(v - (float)h);
          float w   = vv1[q][j];
          __bf16 hw = (__bf16)w;
          h1[q * 4 + j] = hw;
          l1[q * 4 + j] = (__bf16)(w - (float)hw);
        }
      *(v8bf*)&Ahi[r * A_STRIDE + c0]     = h0;
      *(v8bf*)&Ahi[r * A_STRIDE + c0 + 8] = h1;
      *(v8bf*)&Alo[r * A_STRIDE + c0]     = l0;
      *(v8bf*)&Alo[r * A_STRIDE + c0 + 8] = l1;
    }
    __syncthreads();

    // ---- GEMM: (64 x 256) @ (256 x 256), accumulate into acc ----
    const int gw = l_ind[g];
    const __bf16* Bh = WhiT + ((size_t)gw << 16);
    const __bf16* Bl = WloT + ((size_t)gw << 16);
    const int n0 = (wave << 5) + l15;     // this lane's column in N-tile 0
    const int n1 = n0 + 16;               // ... and N-tile 1

    // prefetch next group's weight planes into L2/L0 while we compute
    if (g + 1 < ge) {
      const int gwn = l_ind[g + 1];
      __builtin_prefetch((const char*)(WhiT + ((size_t)gwn << 16)) + tid * 512, 0, 0);
      __builtin_prefetch((const char*)(WloT + ((size_t)gwn << 16)) + tid * 512, 0, 0);
    }

    #pragma unroll
    for (int ks = 0; ks < 8; ++ks) {
      const int kb = ks * 32 + half * 8;  // lane's first contiguous K run

      // B fragments: live across the whole m-tile loop (amortized 4x)
      v16bf Bh0, Bl0, Bh1, Bl1;
      {
        const __bf16* p0 = Bh + (((size_t)n0) << 8) + kb;
        const __bf16* q0 = Bl + (((size_t)n0) << 8) + kb;
        const __bf16* p1 = Bh + (((size_t)n1) << 8) + kb;
        const __bf16* q1 = Bl + (((size_t)n1) << 8) + kb;
        ((v8bf*)&Bh0)[0] = *(const v8bf*)(p0);
        ((v8bf*)&Bh0)[1] = *(const v8bf*)(p0 + 16);
        ((v8bf*)&Bl0)[0] = *(const v8bf*)(q0);
        ((v8bf*)&Bl0)[1] = *(const v8bf*)(q0 + 16);
        ((v8bf*)&Bh1)[0] = *(const v8bf*)(p1);
        ((v8bf*)&Bh1)[1] = *(const v8bf*)(p1 + 16);
        ((v8bf*)&Bl1)[0] = *(const v8bf*)(q1);
        ((v8bf*)&Bl1)[1] = *(const v8bf*)(q1 + 16);
      }

      #pragma unroll
      for (int m = 0; m < MT; ++m) {
        const int arow = m * 16 + l15;
        v16bf Af, Al2;
        ((v8bf*)&Af)[0]  = *(const v8bf*)&Ahi[arow * A_STRIDE + kb];
        ((v8bf*)&Af)[1]  = *(const v8bf*)&Ahi[arow * A_STRIDE + kb + 16];
        ((v8bf*)&Al2)[0] = *(const v8bf*)&Alo[arow * A_STRIDE + kb];
        ((v8bf*)&Al2)[1] = *(const v8bf*)&Alo[arow * A_STRIDE + kb + 16];

        acc[m][0] = __builtin_amdgcn_wmma_f32_16x16x32_bf16(false, Af,  false, Bh0, (short)0, acc[m][0], false, false);
        acc[m][0] = __builtin_amdgcn_wmma_f32_16x16x32_bf16(false, Af,  false, Bl0, (short)0, acc[m][0], false, false);
        acc[m][0] = __builtin_amdgcn_wmma_f32_16x16x32_bf16(false, Al2, false, Bh0, (short)0, acc[m][0], false, false);

        acc[m][1] = __builtin_amdgcn_wmma_f32_16x16x32_bf16(false, Af,  false, Bh1, (short)0, acc[m][1], false, false);
        acc[m][1] = __builtin_amdgcn_wmma_f32_16x16x32_bf16(false, Af,  false, Bl1, (short)0, acc[m][1], false, false);
        acc[m][1] = __builtin_amdgcn_wmma_f32_16x16x32_bf16(false, Al2, false, Bh1, (short)0, acc[m][1], false, false);
      }
    }
    __syncthreads();   // before next group rewrites LDS
  }

  // ---- store: block exclusively owns out[b0:b0+64, Mord, :] — no atomics ---
  // C/D layout: VGPR j -> batch row half*8 + j, column = l15 (+ tile offset)
  #pragma unroll
  for (int m = 0; m < MT; ++m) {
    #pragma unroll
    for (int j = 0; j < 8; ++j) {
      const int bb = b0 + m * 16 + half * 8 + j;
      float* orow = out + (((size_t)bb * NORD + Mord) << 8);
      orow[(wave << 5) + l15]      = acc[m][0][j];
      orow[(wave << 5) + 16 + l15] = acc[m][1][j];
    }
  }
}

// ---------------------------------------------------------------------------
extern "C" void kernel_launch(void* const* d_in, const int* in_sizes, int n_in,
                              void* d_out, int out_size, void* d_ws, size_t ws_size,
                              hipStream_t stream) {
  const float* x   = (const float*)d_in[0];
  const float* sh  = (const float*)d_in[1];
  const float* W   = (const float*)d_in[2];
  const float* CG  = (const float*)d_in[3];
  const int* M1    = (const int*)d_in[4];
  const int* M2    = (const int*)d_in[5];
  const int* seg1  = (const int*)d_in[6];
  const int* l_ind = (const int*)d_in[7];
  const int* seg2  = (const int*)d_in[8];

  const int nnz  = in_sizes[3];
  const int G    = in_sizes[7];
  const int n_w  = in_sizes[2] / (CI * CO);
  const int Btot = in_sizes[0] / (NORD * CI);

  // workspace layout: [0,4KB) index ranges, then bf16 hi/lo weight planes
  int* gStart  = (int*)d_ws;
  int* zStart  = gStart + 32;
  __bf16* WhiT = (__bf16*)((char*)d_ws + 4096);
  __bf16* WloT = WhiT + (size_t)n_w * CI * CO;

  so3_build_ranges<<<1, 256, 0, stream>>>(seg1, nnz, seg2, G, gStart, zStart);

  const int totalW = n_w * CI * CO;
  so3_split_weight<<<(totalW + 255) / 256, 256, 0, stream>>>(W, totalW, WhiT, WloT);

  dim3 grid(Btot / BT, NORD);
  so3_fused<<<grid, 256, 0, stream>>>(x, sh, CG, M1, M2, l_ind,
                                      gStart, zStart, WhiT, WloT, (float*)d_out);
}
